// APSA_24275155157659
// MI455X (gfx1250) — matrix operand order repaired and checked by reading, not compiled, not run on hardware
//
#include <hip/hip_runtime.h>
#include <hip/hip_bf16.h>

// ---------------------------------------------------------------------------
// Problem constants (from reference)
// ---------------------------------------------------------------------------
#define BB 16      // batch
#define CC 16      // channels
#define SS 256     // spatial
#define HH 8       // heads
#define WN 16      // windows per side
#define NW 256     // windows total
#define NP 16      // patches per window
#define PD 256     // patch dim
#define WD 256     // pooled window dim
#define WOD 4096   // window output dim
#define DH 32      // dim per head
#define SCALE 0.17677669529663687f  // 32^-0.5

typedef __attribute__((ext_vector_type(16))) __bf16 bf16x16;
typedef __attribute__((ext_vector_type(8)))  __bf16 bf16x8;
typedef __attribute__((ext_vector_type(8)))  float  f32x8;

// ---------------------------------------------------------------------------
// Reduction helpers (wave32)
// ---------------------------------------------------------------------------
__device__ __forceinline__ float wave_sum(float v) {
#pragma unroll
  for (int o = 16; o > 0; o >>= 1) v += __shfl_xor(v, o, 32);
  return v;
}
__device__ __forceinline__ float wave_max(float v) {
#pragma unroll
  for (int o = 16; o > 0; o >>= 1) v = fmaxf(v, __shfl_xor(v, o, 32));
  return v;
}
__device__ __forceinline__ float block_sum(float v, float* red, int nwaves) {
  v = wave_sum(v);
  int w = threadIdx.x >> 5;
  if ((threadIdx.x & 31) == 0) red[w] = v;
  __syncthreads();
  float s = 0.f;
  for (int i = 0; i < nwaves; ++i) s += red[i];
  __syncthreads();
  return s;
}
__device__ __forceinline__ float block_max(float v, float* red, int nwaves) {
  v = wave_max(v);
  int w = threadIdx.x >> 5;
  if ((threadIdx.x & 31) == 0) red[w] = v;
  __syncthreads();
  float s = -3.4e38f;
  for (int i = 0; i < nwaves; ++i) s = fmaxf(s, red[i]);
  __syncthreads();
  return s;
}

// ---------------------------------------------------------------------------
// Fragment loaders (CDNA5 ISA 7.12.2 wave32 layouts), 128-bit vector loads
// A 16x32: lane row = r; half 0 holds K 0-7 / 16-23, half 1 holds K 8-15 / 24-31
// B 32x16 (from W[N,K] row-major): lane col = r; half 0 K 0-7 / 16-23, half 1 K 8-15 / 24-31
// ---------------------------------------------------------------------------
__device__ __forceinline__ bf16x16 ld_frag_a(const __bf16* p) {
  bf16x8 lo = *(const bf16x8*)(p);
  bf16x8 hi = *(const bf16x8*)(p + 16);
  return __builtin_shufflevector(lo, hi, 0, 1, 2, 3, 4, 5, 6, 7, 8, 9, 10, 11, 12, 13, 14, 15);
}
__device__ __forceinline__ bf16x16 ld_frag_bt(const __bf16* p) {
  bf16x8 lo = *(const bf16x8*)(p);
  bf16x8 hi = *(const bf16x8*)(p + 16);
  return __builtin_shufflevector(lo, hi, 0, 1, 2, 3, 4, 5, 6, 7, 8, 9, 10, 11, 12, 13, 14, 15);
}

// ---------------------------------------------------------------------------
// fp32 -> bf16 conversion (weights)
// ---------------------------------------------------------------------------
__global__ void k_cvt_bf16(const float* __restrict__ src, __bf16* __restrict__ dst, int n) {
  int i = blockIdx.x * blockDim.x + threadIdx.x;
  if (i < n) dst[i] = (__bf16)src[i];
}

// ---------------------------------------------------------------------------
// bf16 WMMA GEMM:  C[M,N] = A[M,K] @ B^T   (B row-major [N,K])
// One wave computes a 16x64 strip (4 accumulators; A fragment reused 4x).
// Software-pipelined: fragments for K-step k+1 are loaded before the 4-WMMA
// chain of step k, so WMMAs never wait on their own loads (double-buffered).
// OUT_MODE: 0 = f32, 1 = bf16, 2 = bf16 batched-transpose (wv_t[b][4096][256],
//           batch = 256 rows, N must be 4096)
// ---------------------------------------------------------------------------
template <int OUT_MODE>
__global__ void k_gemm(const __bf16* __restrict__ A, const __bf16* __restrict__ Bm,
                       void* __restrict__ Cv, int M, int N, int K,
                       int lda, int ldb, int ldc) {
  const int wave = threadIdx.x >> 5;
  const int lane = threadIdx.x & 31;
  const int strips_n = N >> 6;
  const int strip = blockIdx.x * (blockDim.x >> 5) + wave;
  const int total = (M >> 4) * strips_n;
  if (strip >= total) return;
  const int tm = (strip / strips_n) << 4;
  const int tn0 = (strip % strips_n) << 6;
  const int half = lane >> 4;
  const int r = lane & 15;

  f32x8 acc[4] = {};
  const __bf16* ap = A + (size_t)(tm + r) * lda + half * 8;
  const __bf16* bp = Bm + (size_t)(tn0 + r) * ldb + half * 8;

  // prologue: load K-step 0
  bf16x16 a  = ld_frag_a(ap);
  bf16x16 b0 = ld_frag_bt(bp + (size_t)0 * 16 * ldb);
  bf16x16 b1 = ld_frag_bt(bp + (size_t)1 * 16 * ldb);
  bf16x16 b2 = ld_frag_bt(bp + (size_t)2 * 16 * ldb);
  bf16x16 b3 = ld_frag_bt(bp + (size_t)3 * 16 * ldb);
  for (int k0 = 32; k0 < K; k0 += 32) {
    __builtin_prefetch(ap + k0 + 32, 0, 0);  // A K-step k+2 -> global_prefetch
    // issue next K-step loads first ...
    bf16x16 an  = ld_frag_a(ap + k0);
    bf16x16 bn0 = ld_frag_bt(bp + (size_t)0 * 16 * ldb + k0);
    bf16x16 bn1 = ld_frag_bt(bp + (size_t)1 * 16 * ldb + k0);
    bf16x16 bn2 = ld_frag_bt(bp + (size_t)2 * 16 * ldb + k0);
    bf16x16 bn3 = ld_frag_bt(bp + (size_t)3 * 16 * ldb + k0);
    // ... then consume the previous K-step's fragments
    acc[0] = __builtin_amdgcn_wmma_f32_16x16x32_bf16(false, a, false, b0, (short)0,
                                                     acc[0], false, false);
    acc[1] = __builtin_amdgcn_wmma_f32_16x16x32_bf16(false, a, false, b1, (short)0,
                                                     acc[1], false, false);
    acc[2] = __builtin_amdgcn_wmma_f32_16x16x32_bf16(false, a, false, b2, (short)0,
                                                     acc[2], false, false);
    acc[3] = __builtin_amdgcn_wmma_f32_16x16x32_bf16(false, a, false, b3, (short)0,
                                                     acc[3], false, false);
    a = an; b0 = bn0; b1 = bn1; b2 = bn2; b3 = bn3;
  }
  // epilogue: last K-step
  acc[0] = __builtin_amdgcn_wmma_f32_16x16x32_bf16(false, a, false, b0, (short)0,
                                                   acc[0], false, false);
  acc[1] = __builtin_amdgcn_wmma_f32_16x16x32_bf16(false, a, false, b1, (short)0,
                                                   acc[1], false, false);
  acc[2] = __builtin_amdgcn_wmma_f32_16x16x32_bf16(false, a, false, b2, (short)0,
                                                   acc[2], false, false);
  acc[3] = __builtin_amdgcn_wmma_f32_16x16x32_bf16(false, a, false, b3, (short)0,
                                                   acc[3], false, false);

  // C layout: VGPR v -> row tm + half*8 + v, col tn0 + s*16 + r
#pragma unroll
  for (int s = 0; s < 4; ++s) {
#pragma unroll
    for (int v = 0; v < 8; ++v) {
      int row = tm + half * 8 + v;
      int col = tn0 + s * 16 + r;
      if (OUT_MODE == 0) {
        ((float*)Cv)[(size_t)row * ldc + col] = acc[s][v];
      } else if (OUT_MODE == 1) {
        ((__bf16*)Cv)[(size_t)row * ldc + col] = (__bf16)acc[s][v];
      } else {
        // wv_t[b][col][row%256], b = row/256, batch stride 4096*256 = 1<<20
        ((__bf16*)Cv)[((size_t)(row >> 8) << 20) + (size_t)col * 256 + (row & 255)] =
            (__bf16)acc[s][v];
      }
    }
  }
}

// ---------------------------------------------------------------------------
// Patch gather + LayerNorm(PD=256) -> bf16 pf [B*NW*NP, 256]
// ---------------------------------------------------------------------------
__global__ void k_patch_ln(const float* __restrict__ x, const float* __restrict__ g,
                           const float* __restrict__ bta, __bf16* __restrict__ pf) {
  __shared__ float red[8];
  const int row = blockIdx.x;
  const int t = threadIdx.x;
  const int b = row >> 12;
  const int w = (row >> 4) & 255;
  const int n = row & 15;
  const int wy = w >> 4, wx = w & 15, py = n >> 2, px = n & 3;
  const int p1 = t >> 6, p2 = (t >> 4) & 3, c = t & 15;
  const int sy = wy * 16 + py * 4 + p1;
  const int sx = wx * 16 + px * 4 + p2;
  float v = x[(((size_t)b * CC + c) * SS + sy) * SS + sx];
  float m = block_sum(v, red, 8) * (1.0f / PD);
  float d = v - m;
  float var = block_sum(d * d, red, 8) * (1.0f / PD);
  float rs = rsqrtf(var + 1e-5f);
  pf[(size_t)row * PD + t] = (__bf16)(d * rs * g[t] + bta[t]);
}

// ---------------------------------------------------------------------------
// Patch-level attention per (b, window)
// ---------------------------------------------------------------------------
__global__ void k_patch_attn(const __bf16* __restrict__ qk, const __bf16* __restrict__ pv,
                             float* __restrict__ feat) {
  __shared__ float sq[NP * 512];
  __shared__ float sv[NP * 256];
  __shared__ float sc[HH * NP * NP];
  const int bw = blockIdx.x;
  const int t = threadIdx.x;
  for (int it = 0; it < 32; ++it) {
    int e = it * 256 + t;
    sq[e] = (float)qk[(size_t)(bw * NP + (e >> 9)) * 512 + (e & 511)];
  }
  for (int it = 0; it < 16; ++it) {
    int e = it * 256 + t;
    sv[e] = (float)pv[(size_t)(bw * NP + (e >> 8)) * 256 + (e & 255)];
  }
  __syncthreads();
  for (int it = 0; it < 8; ++it) {
    int idx = it * 256 + t;
    int h = idx >> 8, i = (idx >> 4) & 15, j = idx & 15;
    float s = 0.f;
#pragma unroll
    for (int d = 0; d < DH; ++d)
      s += sq[i * 512 + h * DH + d] * sq[j * 512 + 256 + h * DH + d];
    sc[idx] = s * SCALE;
  }
  __syncthreads();
  if (t < HH * NP) {
    float* rowp = &sc[t * NP];
    float m = -3.4e38f;
#pragma unroll
    for (int j = 0; j < NP; ++j) m = fmaxf(m, rowp[j]);
    float s = 0.f;
#pragma unroll
    for (int j = 0; j < NP; ++j) { rowp[j] = __expf(rowp[j] - m); s += rowp[j]; }
    float inv = 1.0f / s;
#pragma unroll
    for (int j = 0; j < NP; ++j) rowp[j] *= inv;
  }
  __syncthreads();
  for (int it = 0; it < 16; ++it) {
    int idx = it * 256 + t;
    int i = idx >> 8, h = (idx >> 5) & 7, d = idx & 31;
    float s = 0.f;
#pragma unroll
    for (int j = 0; j < NP; ++j)
      s += sc[(h * NP + i) * NP + j] * sv[j * 256 + h * DH + d];
    feat[(size_t)bw * WOD + idx] = s;
  }
}

// ---------------------------------------------------------------------------
// 4x4 average pool + gather + LayerNorm(WD=256) -> bf16 wf [B*NW, 256]
// ---------------------------------------------------------------------------
__global__ void k_pool_ln(const float* __restrict__ x, const float* __restrict__ g,
                          const float* __restrict__ bta, __bf16* __restrict__ wf) {
  __shared__ float red[8];
  const int row = blockIdx.x;
  const int t = threadIdx.x;
  const int b = row >> 8, w = row & 255;
  const int wy = w >> 4, wx = w & 15;
  const int ky = t >> 6, kx = (t >> 4) & 3, c = t & 15;
  const int pi = wy * 4 + ky, pj = wx * 4 + kx;
  float s = 0.f;
#pragma unroll
  for (int a = 0; a < 4; ++a)
#pragma unroll
    for (int e = 0; e < 4; ++e)
      s += x[(((size_t)b * CC + c) * SS + pi * 4 + a) * SS + pj * 4 + e];
  float v = s * (1.0f / 16.0f);
  float m = block_sum(v, red, 8) * (1.0f / WD);
  float d = v - m;
  float var = block_sum(d * d, red, 8) * (1.0f / WD);
  float rs = rsqrtf(var + 1e-5f);
  wf[(size_t)row * WD + t] = (__bf16)(d * rs * g[t] + bta[t]);
}

// ---------------------------------------------------------------------------
// Window attention weights: one block per (b,h,n) row; softmax over 256 cols
// ---------------------------------------------------------------------------
__global__ void k_win_attn(const float* __restrict__ qkw, float* __restrict__ ww) {
  __shared__ float sq[DH];
  __shared__ float red[8];
  const int bid = blockIdx.x;
  const int b = bid >> 11, h = (bid >> 8) & 7, n = bid & 255;
  const int j = threadIdx.x;
  if (j < DH) sq[j] = qkw[((size_t)b * NW + n) * 512 + h * DH + j];
  __syncthreads();
  float s = 0.f;
#pragma unroll
  for (int d = 0; d < DH; ++d)
    s += sq[d] * qkw[((size_t)b * NW + j) * 512 + 256 + h * DH + d];
  s *= SCALE;
  float m = block_max(s, red, 8);
  float e = __expf(s - m);
  float sum = block_sum(e, red, 8);
  ww[(((size_t)b * HH + h) * NW + n) * NW + j] = e / sum;
}

// ---------------------------------------------------------------------------
// Diagonal reweighting: diag_s f32 [B*H*NW], ndw bf16 [B*H, NW, NW]
// ---------------------------------------------------------------------------
__global__ void k_diag(const float* __restrict__ ww, float* __restrict__ diag_s,
                       __bf16* __restrict__ ndw) {
  const int bh = blockIdx.x;
  const int n = threadIdx.x;
  const float* base = ww + (size_t)bh * NW * NW;
  float diag = base[n * NW + n];
  bool li = diag < 0.5f;
  diag_s[bh * NW + n] = li ? 0.5f : diag;
  float f = li ? 0.5f / (1.0f - diag) : 1.0f;
  for (int j = 0; j < NW; ++j) {
    float v = (j == n) ? 0.f : base[n * NW + j] * f;
    ndw[(size_t)bh * NW * NW + n * NW + j] = (__bf16)v;
  }
}

// ---------------------------------------------------------------------------
// Window gather + LayerNorm(WOD=4096) -> bf16 wflat [B*NW, 4096]
// ---------------------------------------------------------------------------
__global__ void k_wflat_ln(const float* __restrict__ x, const float* __restrict__ g,
                           const float* __restrict__ bta, __bf16* __restrict__ wflat) {
  __shared__ float red[8];
  const int row = blockIdx.x;
  const int t = threadIdx.x;
  const int b = row >> 8, w = row & 255;
  const int wy = w >> 4, wx = w & 15;
  float vals[16];
  float s = 0.f;
#pragma unroll
  for (int i = 0; i < 16; ++i) {
    int idx = i * 256 + t;
    int syi = idx >> 8, sxi = (idx >> 4) & 15, c = idx & 15;
    float v = x[(((size_t)b * CC + c) * SS + wy * 16 + syi) * SS + wx * 16 + sxi];
    vals[i] = v;
    s += v;
  }
  float m = block_sum(s, red, 8) * (1.0f / WOD);
  float vs = 0.f;
#pragma unroll
  for (int i = 0; i < 16; ++i) { float d = vals[i] - m; vs += d * d; }
  float var = block_sum(vs, red, 8) * (1.0f / WOD);
  float rs = rsqrtf(var + 1e-5f);
#pragma unroll
  for (int i = 0; i < 16; ++i) {
    int idx = i * 256 + t;
    wflat[(size_t)row * WOD + idx] = (__bf16)((vals[i] - m) * rs * g[idx] + bta[idx]);
  }
}

// ---------------------------------------------------------------------------
// wfeat[b,h] = ndw[b,h] @ wv[b,h] + diag_s * feat   (WMMA, BT path via wv_t)
// wv_t: [b*H+h][512][256] bf16 (d-major, window minor) -> B^T layout, ldb=256
// out wfeat bf16 [b, NW, 4096], col = h*512 + d.  Software-pipelined like k_gemm.
// ---------------------------------------------------------------------------
__global__ void k_gemm_apply(const __bf16* __restrict__ ndw, const __bf16* __restrict__ wv_t,
                             const float* __restrict__ diag_s, const float* __restrict__ feat,
                             __bf16* __restrict__ wfeat) {
  const int bh = blockIdx.y;
  const int b = bh >> 3, h = bh & 7;
  const int wave = threadIdx.x >> 5;
  const int lane = threadIdx.x & 31;
  const int strips_n = 512 >> 6;  // 8
  const int strip = blockIdx.x * (blockDim.x >> 5) + wave;
  if (strip >= 16 * strips_n) return;
  const int tm = (strip / strips_n) << 4;
  const int tn0 = (strip % strips_n) << 6;
  const int half = lane >> 4;
  const int r = lane & 15;

  const __bf16* A = ndw + (size_t)bh * NW * NW;        // [256,256]
  const __bf16* Bm = wv_t + (size_t)bh * 512 * 256;    // [512,256] = W[N,K]

  f32x8 acc[4] = {};
  const __bf16* ap = A + (size_t)(tm + r) * NW + half * 8;
  const __bf16* bp = Bm + (size_t)(tn0 + r) * 256 + half * 8;

  bf16x16 a  = ld_frag_a(ap);
  bf16x16 b0 = ld_frag_bt(bp + (size_t)0 * 16 * 256);
  bf16x16 b1 = ld_frag_bt(bp + (size_t)1 * 16 * 256);
  bf16x16 b2 = ld_frag_bt(bp + (size_t)2 * 16 * 256);
  bf16x16 b3 = ld_frag_bt(bp + (size_t)3 * 16 * 256);
  for (int k0 = 32; k0 < NW; k0 += 32) {
    bf16x16 an  = ld_frag_a(ap + k0);
    bf16x16 bn0 = ld_frag_bt(bp + (size_t)0 * 16 * 256 + k0);
    bf16x16 bn1 = ld_frag_bt(bp + (size_t)1 * 16 * 256 + k0);
    bf16x16 bn2 = ld_frag_bt(bp + (size_t)2 * 16 * 256 + k0);
    bf16x16 bn3 = ld_frag_bt(bp + (size_t)3 * 16 * 256 + k0);
    acc[0] = __builtin_amdgcn_wmma_f32_16x16x32_bf16(false, a, false, b0, (short)0,
                                                     acc[0], false, false);
    acc[1] = __builtin_amdgcn_wmma_f32_16x16x32_bf16(false, a, false, b1, (short)0,
                                                     acc[1], false, false);
    acc[2] = __builtin_amdgcn_wmma_f32_16x16x32_bf16(false, a, false, b2, (short)0,
                                                     acc[2], false, false);
    acc[3] = __builtin_amdgcn_wmma_f32_16x16x32_bf16(false, a, false, b3, (short)0,
                                                     acc[3], false, false);
    a = an; b0 = bn0; b1 = bn1; b2 = bn2; b3 = bn3;
  }
  acc[0] = __builtin_amdgcn_wmma_f32_16x16x32_bf16(false, a, false, b0, (short)0,
                                                   acc[0], false, false);
  acc[1] = __builtin_amdgcn_wmma_f32_16x16x32_bf16(false, a, false, b1, (short)0,
                                                   acc[1], false, false);
  acc[2] = __builtin_amdgcn_wmma_f32_16x16x32_bf16(false, a, false, b2, (short)0,
                                                   acc[2], false, false);
  acc[3] = __builtin_amdgcn_wmma_f32_16x16x32_bf16(false, a, false, b3, (short)0,
                                                   acc[3], false, false);

#pragma unroll
  for (int s = 0; s < 4; ++s) {
#pragma unroll
    for (int v = 0; v < 8; ++v) {
      int mrow = tm + half * 8 + v;
      size_t col = (size_t)h * 512 + tn0 + s * 16 + r;
      float val = acc[s][v] +
                  diag_s[bh * NW + mrow] * feat[((size_t)b * NW + mrow) * WOD + col];
      wfeat[((size_t)b * NW + mrow) * WOD + col] = (__bf16)val;
    }
  }
}

// ---------------------------------------------------------------------------
// Scatter out_lin [B*NW, 4096] back to [B,C,S,S] with bias + residual
// ---------------------------------------------------------------------------
__global__ void k_scatter(const float* __restrict__ out_lin, const float* __restrict__ bo,
                          const float* __restrict__ x, float* __restrict__ out) {
  const int row = blockIdx.x;
  const int t = threadIdx.x;
  const int b = row >> 8, w = row & 255;
  const int wy = w >> 4, wx = w & 15;
#pragma unroll
  for (int i = 0; i < 16; ++i) {
    int idx = i * 256 + t;
    int syi = idx >> 8, sxi = (idx >> 4) & 15, c = idx & 15;
    size_t xi = (((size_t)b * CC + c) * SS + wy * 16 + syi) * SS + wx * 16 + sxi;
    out[xi] = out_lin[(size_t)row * WOD + idx] + bo[idx] + x[xi];
  }
}

// ---------------------------------------------------------------------------
// Host launcher
// ---------------------------------------------------------------------------
extern "C" void kernel_launch(void* const* d_in, const int* in_sizes, int n_in,
                              void* d_out, int out_size, void* d_ws, size_t ws_size,
                              hipStream_t stream) {
  const float* x    = (const float*)d_in[0];
  const float* pn_g = (const float*)d_in[1];
  const float* pn_b = (const float*)d_in[2];
  const float* Wpqk = (const float*)d_in[3];
  const float* Wpv  = (const float*)d_in[4];
  const float* on_g = (const float*)d_in[5];
  const float* on_b = (const float*)d_in[6];
  const float* Wwqk = (const float*)d_in[7];
  const float* wn_g = (const float*)d_in[8];
  const float* wn_b = (const float*)d_in[9];
  const float* Wwv  = (const float*)d_in[10];
  const float* Wo   = (const float*)d_in[11];
  const float* bo   = (const float*)d_in[12];
  float* out = (float*)d_out;

  char* p = (char*)d_ws;
  auto carve = [&](size_t bytes) {
    char* r = p;
    p += (bytes + 255) & ~(size_t)255;
    return (void*)r;
  };
  const size_t ROWS_P = (size_t)BB * NW * NP;  // 65536
  const size_t ROWS_W = (size_t)BB * NW;       // 4096

  __bf16* wpqk_b  = (__bf16*)carve(512 * 256 * 2);
  __bf16* wpv_b   = (__bf16*)carve(256 * 256 * 2);
  __bf16* wwqk_b  = (__bf16*)carve(512 * 256 * 2);
  __bf16* wwv_b   = (__bf16*)carve((size_t)WOD * WOD * 2);
  __bf16* wo_b    = (__bf16*)carve((size_t)WOD * WOD * 2);
  __bf16* pf_b    = (__bf16*)carve(ROWS_P * PD * 2);
  __bf16* qk_p    = (__bf16*)carve(ROWS_P * 512 * 2);
  __bf16* pv_p    = (__bf16*)carve(ROWS_P * 256 * 2);
  float*  feat    = (float*)carve(ROWS_W * WOD * 4);
  __bf16* wf_b    = (__bf16*)carve(ROWS_W * WD * 2);
  float*  qkw     = (float*)carve(ROWS_W * 512 * 4);
  float*  ww      = (float*)carve((size_t)BB * HH * NW * NW * 4);
  float*  diag_s  = (float*)carve((size_t)BB * HH * NW * 4);
  __bf16* ndw_b   = (__bf16*)carve((size_t)BB * HH * NW * NW * 2);
  __bf16* wflat_b = (__bf16*)carve(ROWS_W * WOD * 2);
  __bf16* wv_t    = (__bf16*)carve(ROWS_W * WOD * 2);  // [b][4096][256]
  __bf16* wfeat_b = (__bf16*)carve(ROWS_W * WOD * 2);
  float*  out_lin = (float*)carve(ROWS_W * WOD * 4);
  (void)ws_size; (void)n_in; (void)in_sizes; (void)out_size;

  auto cvt = [&](const float* s, __bf16* d, int n) {
    k_cvt_bf16<<<(n + 255) / 256, 256, 0, stream>>>(s, d, n);
  };
  cvt(Wpqk, wpqk_b, 512 * 256);
  cvt(Wpv,  wpv_b,  256 * 256);
  cvt(Wwqk, wwqk_b, 512 * 256);
  cvt(Wwv,  wwv_b,  WOD * WOD);
  cvt(Wo,   wo_b,   WOD * WOD);

  // patch LN
  k_patch_ln<<<(int)ROWS_P, 256, 0, stream>>>(x, pn_g, pn_b, pf_b);

  // patch GEMMs (strips of 16x64, 4 waves/block of 128 threads)
  k_gemm<1><<<(65536 / 16) * (512 / 64) / 4, 128, 0, stream>>>(
      pf_b, wpqk_b, qk_p, 65536, 512, 256, 256, 256, 512);
  k_gemm<1><<<(65536 / 16) * (256 / 64) / 4, 128, 0, stream>>>(
      pf_b, wpv_b, pv_p, 65536, 256, 256, 256, 256, 256);

  // patch attention
  k_patch_attn<<<(int)ROWS_W, 256, 0, stream>>>(qk_p, pv_p, feat);

  // window path
  k_pool_ln<<<(int)ROWS_W, 256, 0, stream>>>(x, on_g, on_b, wf_b);
  k_gemm<0><<<(4096 / 16) * (512 / 64) / 4, 128, 0, stream>>>(
      wf_b, wwqk_b, qkw, 4096, 512, 256, 256, 256, 512);
  k_win_attn<<<BB * HH * NW, 256, 0, stream>>>(qkw, ww);
  k_diag<<<BB * HH, 256, 0, stream>>>(ww, diag_s, ndw_b);

  // wflat LN + big GEMM wv (stored batch-transposed for the apply GEMM)
  k_wflat_ln<<<(int)ROWS_W, 256, 0, stream>>>(x, wn_g, wn_b, wflat_b);
  k_gemm<2><<<(4096 / 16) * (4096 / 64) / 4, 128, 0, stream>>>(
      wflat_b, wwv_b, wv_t, 4096, 4096, 4096, 4096, 4096, 4096);

  // wfeat = ndw @ wv + diag_s * feat
  {
    dim3 grid(16 * 8 / 4, BB * HH);
    k_gemm_apply<<<grid, 128, 0, stream>>>(ndw_b, wv_t, diag_s, feat, wfeat_b);
  }

  // out = wfeat @ Wo^T, then scatter + bias + residual
  k_gemm<0><<<(4096 / 16) * (4096 / 64) / 4, 128, 0, stream>>>(
      wfeat_b, wo_b, out_lin, 4096, 4096, 4096, 4096, 4096, 4096);
  k_scatter<<<(int)ROWS_W, 256, 0, stream>>>(out_lin, bo, x, out);
}
